// EncoderDecoderModel_56581899157950
// MI455X (gfx1250) — matrix operand order repaired and checked by reading, not compiled, not run on hardware
//
#include <hip/hip_runtime.h>
#include <math.h>

typedef float  v8f  __attribute__((ext_vector_type(8)));
typedef __bf16 v16bf __attribute__((ext_vector_type(16)));
typedef __bf16 v8bf  __attribute__((ext_vector_type(8)));

#define B_    32
#define TSEQ  64
#define E_    1024
#define H_    1024
#define G4H   4096
#define VOC   32000

__device__ __forceinline__ float sigmoidf_(float x) { return 1.0f / (1.0f + expf(-x)); }

__device__ __forceinline__ v16bf cat8(v8bf a, v8bf b) {
  return __builtin_shufflevector(a, b, 0,1,2,3,4,5,6,7,8,9,10,11,12,13,14,15);
}

// split 8 consecutive f32 into bf16 hi + bf16 lo (residual)
__device__ __forceinline__ void split8(const float* __restrict__ p, v8bf& h, v8bf& l) {
  v8f x = *(const v8f*)p;
#pragma unroll
  for (int i = 0; i < 8; ++i) {
    float xi = x[i];
    __bf16 hb = (__bf16)xi;
    h[i] = hb;
    l[i] = (__bf16)(xi - (float)hb);
  }
}

#define WMMA_BF16(A, Bm, Cacc) \
  __builtin_amdgcn_wmma_f32_16x16x32_bf16(false, (A), false, (Bm), (short)0, (Cacc), false, false)

// ---------------------------------------------------------------------------
// Weight pre-split: whole-array f32 -> bf16 hi + bf16 lo
// ---------------------------------------------------------------------------
__global__ void split_w(const float* __restrict__ w, __bf16* __restrict__ whi,
                        __bf16* __restrict__ wlo, int n) {
  int i = blockIdx.x * blockDim.x + threadIdx.x;
  if (i < n) {
    float x = w[i];
    __bf16 h = (__bf16)x;
    whi[i] = h;
    wlo[i] = (__bf16)(x - (float)h);
  }
}

// ---------------------------------------------------------------------------
// Embedding gather + hi/lo split: out[(t*B+b)] = split(emb[idx[b*T+t]])
// ---------------------------------------------------------------------------
__global__ void embed_rows_hl(const float* __restrict__ emb, const int* __restrict__ idx,
                              __bf16* __restrict__ ohi, __bf16* __restrict__ olo) {
  int m = blockIdx.x;             // 0 .. T*B-1
  int t = m / B_, b = m % B_;
  int tok = idx[b * TSEQ + t];
  const float* src = emb + (size_t)tok * E_;
  __bf16* dh = ohi + (size_t)m * E_;
  __bf16* dl = olo + (size_t)m * E_;
  for (int e = threadIdx.x; e < E_; e += blockDim.x) {
    float x = src[e];
    __bf16 h = (__bf16)x;
    dh[e] = h;
    dl[e] = (__bf16)(x - (float)h);
  }
}

// ---------------------------------------------------------------------------
// C[m,n] = sum_k A[m,k]*W[n,k] + bias[n] via bf16x3 WMMA (16x16x32).
// A pre-split hi/lo bf16 (lda=K). PRE=true: W pre-split bf16 hi/lo.
// PRE=false: W is f32, split in-register. Block = 8 waves, tile 64x256.
// ---------------------------------------------------------------------------
template <bool PRE>
__global__ __launch_bounds__(256) void gemm_bf16x3(
    const __bf16* __restrict__ Ahi, const __bf16* __restrict__ Alo,
    const float* __restrict__ Wf,
    const __bf16* __restrict__ Whi, const __bf16* __restrict__ Wlo,
    const float* __restrict__ bias, float* __restrict__ C, int K, int ldc) {
  const int lane = threadIdx.x & 31;
  const int wid  = threadIdx.x >> 5;
  const int wm = wid & 1, wn = wid >> 1;
  const int m0 = blockIdx.y * 64 + wm * 32;
  const int n0 = blockIdx.x * 256 + wn * 64;
  const int hi = lane >> 4, lo = lane & 15;
  const int aoff0 = hi * 16, aoff1 = hi * 16 + 8;  // A-frag K runs
  const int boff  = hi * 16;                       // B-frag K run

  const __bf16* ah[2] = { Ahi + (size_t)(m0 + lo) * K, Ahi + (size_t)(m0 + 16 + lo) * K };
  const __bf16* al[2] = { Alo + (size_t)(m0 + lo) * K, Alo + (size_t)(m0 + 16 + lo) * K };
  const float*  wpf[4];
  const __bf16* wph[4];
  const __bf16* wpl[4];
#pragma unroll
  for (int j = 0; j < 4; ++j) {
    size_t row = (size_t)(n0 + j * 16 + lo) * K;
    if (PRE) { wph[j] = Whi + row; wpl[j] = Wlo + row; }
    else     { wpf[j] = Wf + row; }
  }

  v8f acc[2][4];
#pragma unroll
  for (int f = 0; f < 2; ++f)
#pragma unroll
    for (int j = 0; j < 4; ++j)
      acc[f][j] = (v8f){0.f,0.f,0.f,0.f,0.f,0.f,0.f,0.f};

  for (int k = 0; k < K; k += 32) {
    v16bf a_hi[2], a_lo[2];
#pragma unroll
    for (int f = 0; f < 2; ++f) {
      a_hi[f] = cat8(*(const v8bf*)(ah[f] + k + aoff0), *(const v8bf*)(ah[f] + k + aoff1));
      a_lo[f] = cat8(*(const v8bf*)(al[f] + k + aoff0), *(const v8bf*)(al[f] + k + aoff1));
    }
    v16bf b_hi[4], b_lo[4];
#pragma unroll
    for (int j = 0; j < 4; ++j) {
      if (PRE) {
        b_hi[j] = cat8(*(const v8bf*)(wph[j] + k + boff), *(const v8bf*)(wph[j] + k + boff + 8));
        b_lo[j] = cat8(*(const v8bf*)(wpl[j] + k + boff), *(const v8bf*)(wpl[j] + k + boff + 8));
      } else {
        v8bf h0, l0, h1, l1;
        split8(wpf[j] + k + boff,     h0, l0);
        split8(wpf[j] + k + boff + 8, h1, l1);
        b_hi[j] = cat8(h0, h1);
        b_lo[j] = cat8(l0, l1);
      }
    }
#pragma unroll
    for (int f = 0; f < 2; ++f)
#pragma unroll
      for (int j = 0; j < 4; ++j) {
        acc[f][j] = WMMA_BF16(a_lo[f], b_hi[j], acc[f][j]);
        acc[f][j] = WMMA_BF16(a_hi[f], b_lo[j], acc[f][j]);
        acc[f][j] = WMMA_BF16(a_hi[f], b_hi[j], acc[f][j]);
      }
  }

  // C/D layout: VGPR r -> M = r + 8*hi, N = lo (within 16x16 frag)
#pragma unroll
  for (int f = 0; f < 2; ++f)
#pragma unroll
    for (int j = 0; j < 4; ++j) {
      int n = n0 + j * 16 + lo;
      float bs = bias[n];
#pragma unroll
      for (int r = 0; r < 8; ++r) {
        int m = m0 + f * 16 + r + hi * 8;
        C[(size_t)m * ldc + n] = acc[f][j][r] + bs;
      }
    }
}

// ---------------------------------------------------------------------------
// One LSTM step with bf16x3 recurrent GEMM. Block owns a 16-wide h slice.
// K=1024 split over 8 waves, LDS reduce, then cell update.
// PRE=true: Whh pre-split bf16 hi/lo; else split f32 in-register.
// ---------------------------------------------------------------------------
template <bool PRE>
__global__ __launch_bounds__(256) void lstm_step_bf16x3(
    const __bf16* __restrict__ hprev_hi, const __bf16* __restrict__ hprev_lo, int hprev_stride,
    const float* __restrict__ cprev,
    const float* __restrict__ Whhf,
    const __bf16* __restrict__ Whhh, const __bf16* __restrict__ Whhl,
    const float* __restrict__ bhh,
    const float* __restrict__ xg,          // [B][4H] for this timestep
    float* __restrict__ hout, __bf16* __restrict__ hout_hi, __bf16* __restrict__ hout_lo,
    int hout_stride, float* __restrict__ cout) {
  __shared__ float red[8 * 8 * 32 * 8];    // [wave][acc][lane][r] = 64 KB
  const int lane = threadIdx.x & 31;
  const int w = threadIdx.x >> 5;
  const int hi = lane >> 4, lo = lane & 15;
  const int h0 = blockIdx.x * 16;
  const int KW = H_ / 8;                   // 128 K per wave
  const int kb = w * KW;
  const int aoff0 = hi * 16, aoff1 = hi * 16 + 8;
  const int boff  = hi * 16;

  const __bf16* ah[2] = { hprev_hi + (size_t)(lo) * hprev_stride + kb,
                          hprev_hi + (size_t)(16 + lo) * hprev_stride + kb };
  const __bf16* al[2] = { hprev_lo + (size_t)(lo) * hprev_stride + kb,
                          hprev_lo + (size_t)(16 + lo) * hprev_stride + kb };
  const float*  wpf[4];
  const __bf16* wph[4];
  const __bf16* wpl[4];
#pragma unroll
  for (int g = 0; g < 4; ++g) {
    size_t row = (size_t)(g * H_ + h0 + lo) * H_ + kb;
    if (PRE) { wph[g] = Whhh + row; wpl[g] = Whhl + row; }
    else     { wpf[g] = Whhf + row; }
  }

  v8f acc[8];
#pragma unroll
  for (int a = 0; a < 8; ++a) acc[a] = (v8f){0.f,0.f,0.f,0.f,0.f,0.f,0.f,0.f};

  for (int k = 0; k < KW; k += 32) {
    v16bf a_hi[2], a_lo[2];
#pragma unroll
    for (int f = 0; f < 2; ++f) {
      a_hi[f] = cat8(*(const v8bf*)(ah[f] + k + aoff0), *(const v8bf*)(ah[f] + k + aoff1));
      a_lo[f] = cat8(*(const v8bf*)(al[f] + k + aoff0), *(const v8bf*)(al[f] + k + aoff1));
    }
#pragma unroll
    for (int g = 0; g < 4; ++g) {
      v16bf b_hi, b_lo;
      if (PRE) {
        b_hi = cat8(*(const v8bf*)(wph[g] + k + boff), *(const v8bf*)(wph[g] + k + boff + 8));
        b_lo = cat8(*(const v8bf*)(wpl[g] + k + boff), *(const v8bf*)(wpl[g] + k + boff + 8));
      } else {
        v8bf h0v, l0v, h1v, l1v;
        split8(wpf[g] + k + boff,     h0v, l0v);
        split8(wpf[g] + k + boff + 8, h1v, l1v);
        b_hi = cat8(h0v, h1v);
        b_lo = cat8(l0v, l1v);
      }
#pragma unroll
      for (int f = 0; f < 2; ++f) {
        int a = g * 2 + f;
        acc[a] = WMMA_BF16(a_lo[f], b_hi, acc[a]);
        acc[a] = WMMA_BF16(a_hi[f], b_lo, acc[a]);
        acc[a] = WMMA_BF16(a_hi[f], b_hi, acc[a]);
      }
    }
  }

#pragma unroll
  for (int a = 0; a < 8; ++a)
#pragma unroll
    for (int r = 0; r < 8; ++r)
      red[((w * 8 + a) * 32 + lane) * 8 + r] = acc[a][r];
  __syncthreads();

  // pointwise: 32 batches x 16 h -> 512 pairs, 2 per thread
  for (int p = threadIdx.x; p < B_ * 16; p += 256) {
    int b = p >> 4;
    int hl = p & 15;
    int f = b >> 4;
    int r = b & 7;
    int ln = ((b >> 3) & 1) * 16 + hl;
    float gi = 0.f, gf = 0.f, gg = 0.f, go = 0.f;
    for (int ww = 0; ww < 8; ++ww) {
      gi += red[((ww * 8 + 0 + f) * 32 + ln) * 8 + r];
      gf += red[((ww * 8 + 2 + f) * 32 + ln) * 8 + r];
      gg += red[((ww * 8 + 4 + f) * 32 + ln) * 8 + r];
      go += red[((ww * 8 + 6 + f) * 32 + ln) * 8 + r];
    }
    int h = h0 + hl;
    gi += xg[(size_t)b * G4H + 0 * H_ + h] + bhh[0 * H_ + h];
    gf += xg[(size_t)b * G4H + 1 * H_ + h] + bhh[1 * H_ + h];
    gg += xg[(size_t)b * G4H + 2 * H_ + h] + bhh[2 * H_ + h];
    go += xg[(size_t)b * G4H + 3 * H_ + h] + bhh[3 * H_ + h];
    float c  = sigmoidf_(gf) * cprev[b * H_ + h] + sigmoidf_(gi) * tanhf(gg);
    float hv = sigmoidf_(go) * tanhf(c);
    cout[b * H_ + h] = c;
    size_t ho = (size_t)b * hout_stride + h;
    hout[ho] = hv;
    __bf16 hb = (__bf16)hv;
    hout_hi[ho] = hb;
    hout_lo[ho] = (__bf16)(hv - (float)hb);
  }
}

// ---------------------------------------------------------------------------
// Gather encoder state at last valid timestep; emit h as bf16 hi/lo, c as f32
// ---------------------------------------------------------------------------
__global__ void gather_last(const int* __restrict__ mask, const float* __restrict__ hs,
                            const float* __restrict__ cs, __bf16* __restrict__ hinit_hi,
                            __bf16* __restrict__ hinit_lo, float* __restrict__ cinit) {
  int i = blockIdx.x * blockDim.x + threadIdx.x;   // 0 .. B*H-1
  int b = i / H_, h = i % H_;
  int len = 0;
  for (int t = 0; t < TSEQ; ++t) len += mask[b * TSEQ + t];
  int tl = len - 1;
  float hv = hs[((size_t)tl * B_ + b) * H_ + h];
  __bf16 hb = (__bf16)hv;
  hinit_hi[i] = hb;
  hinit_lo[i] = (__bf16)(hv - (float)hb);
  cinit[i] = cs[((size_t)tl * B_ + b) * H_ + h];
}

// ---------------------------------------------------------------------------
// In-place row-wise log_softmax over VOC entries (one block per row)
// ---------------------------------------------------------------------------
__global__ __launch_bounds__(256) void log_softmax_rows(float* __restrict__ logits) {
  __shared__ float s[256];
  float* p = logits + (size_t)blockIdx.x * VOC;
  float mx = -INFINITY;
  for (int v = threadIdx.x; v < VOC; v += 256) mx = fmaxf(mx, p[v]);
  s[threadIdx.x] = mx; __syncthreads();
  for (int st = 128; st > 0; st >>= 1) {
    if (threadIdx.x < st) s[threadIdx.x] = fmaxf(s[threadIdx.x], s[threadIdx.x + st]);
    __syncthreads();
  }
  mx = s[0]; __syncthreads();
  float sum = 0.f;
  for (int v = threadIdx.x; v < VOC; v += 256) sum += expf(p[v] - mx);
  s[threadIdx.x] = sum; __syncthreads();
  for (int st = 128; st > 0; st >>= 1) {
    if (threadIdx.x < st) s[threadIdx.x] += s[threadIdx.x + st];
    __syncthreads();
  }
  float lse = logf(s[0]) + mx;
  for (int v = threadIdx.x; v < VOC; v += 256) p[v] = p[v] - lse;
}

// ---------------------------------------------------------------------------
extern "C" void kernel_launch(void* const* d_in, const int* in_sizes, int n_in,
                              void* d_out, int out_size, void* d_ws, size_t ws_size,
                              hipStream_t stream) {
  (void)in_sizes; (void)n_in; (void)out_size;
  const int*   x      = (const int*)d_in[0];
  const int*   x_mask = (const int*)d_in[1];
  const int*   y      = (const int*)d_in[2];
  const float* h0     = (const float*)d_in[3];
  const float* c0     = (const float*)d_in[4];
  const float* emb_en = (const float*)d_in[5];
  const float* emb_cn = (const float*)d_in[6];
  const float* Wih_e  = (const float*)d_in[7];
  const float* Whh_e  = (const float*)d_in[8];
  const float* bih_e  = (const float*)d_in[9];
  const float* bhh_e  = (const float*)d_in[10];
  const float* Wih_d  = (const float*)d_in[11];
  const float* Whh_d  = (const float*)d_in[12];
  const float* bih_d  = (const float*)d_in[13];
  const float* bhh_d  = (const float*)d_in[14];
  const float* Wout   = (const float*)d_in[15];
  const float* bout   = (const float*)d_in[16];

  float* decoded = (float*)d_out;                          // [B,TY,VOC]
  float* hidd    = decoded + (size_t)B_ * TSEQ * VOC;      // [B,TY,H]

  char* wsb = (char*)d_ws;
  const size_t NE = (size_t)TSEQ * B_ * H_;                // 2M elements
  float*  xg     = (float*)wsb;                wsb += (size_t)TSEQ * B_ * G4H * 4;  // 33.5 MB
  float*  hs_e   = (float*)wsb;                wsb += NE * 4;
  float*  cs_e   = (float*)wsb;                wsb += NE * 4;
  __bf16* xe_hi  = (__bf16*)wsb;               wsb += NE * 2;
  __bf16* xe_lo  = (__bf16*)wsb;               wsb += NE * 2;
  __bf16* hsh_e  = (__bf16*)wsb;               wsb += NE * 2;
  __bf16* hsl_e  = (__bf16*)wsb;               wsb += NE * 2;
  __bf16* hdh    = (__bf16*)wsb;               wsb += NE * 2;
  __bf16* hdl    = (__bf16*)wsb;               wsb += NE * 2;
  __bf16* h0hi   = (__bf16*)wsb;               wsb += (size_t)B_ * H_ * 2;
  __bf16* h0lo   = (__bf16*)wsb;               wsb += (size_t)B_ * H_ * 2;
  __bf16* hinhi  = (__bf16*)wsb;               wsb += (size_t)B_ * H_ * 2;
  __bf16* hinlo  = (__bf16*)wsb;               wsb += (size_t)B_ * H_ * 2;
  float*  cinit  = (float*)wsb;                wsb += (size_t)B_ * H_ * 4;
  float*  cb0    = (float*)wsb;                wsb += (size_t)B_ * H_ * 4;
  float*  cb1    = (float*)wsb;                wsb += (size_t)B_ * H_ * 4;
  // optional pre-split weight buffers (used only if ws_size allows)
  __bf16* wihh   = (__bf16*)wsb;               wsb += (size_t)G4H * E_ * 2;  // 8 MB
  __bf16* wihl   = (__bf16*)wsb;               wsb += (size_t)G4H * E_ * 2;
  __bf16* whhh   = (__bf16*)wsb;               wsb += (size_t)G4H * H_ * 2;
  __bf16* whhl   = (__bf16*)wsb;               wsb += (size_t)G4H * H_ * 2;
  __bf16* wouth  = (__bf16*)wsb;               wsb += (size_t)VOC * H_ * 2;  // 65.5 MB
  __bf16* woutl  = (__bf16*)wsb;               wsb += (size_t)VOC * H_ * 2;
  const bool big = (size_t)(wsb - (char*)d_ws) <= ws_size;

  const int M = TSEQ * B_;   // 2048
  const int NIH = G4H * E_;  // Wih elements
  const int NHH = G4H * H_;  // Whh elements
  const int NOU = VOC * H_;  // Wout elements

  // ---- encoder ----
  split_w<<<(B_ * H_) / 256, 256, 0, stream>>>(h0, h0hi, h0lo, B_ * H_);
  embed_rows_hl<<<M, 256, 0, stream>>>(emb_en, x, xe_hi, xe_lo);
  if (big) {
    split_w<<<NIH / 256, 256, 0, stream>>>(Wih_e, wihh, wihl, NIH);
    split_w<<<NHH / 256, 256, 0, stream>>>(Whh_e, whhh, whhl, NHH);
    gemm_bf16x3<true><<<dim3(G4H / 256, M / 64), 256, 0, stream>>>(
        xe_hi, xe_lo, nullptr, wihh, wihl, bih_e, xg, E_, G4H);
  } else {
    gemm_bf16x3<false><<<dim3(G4H / 256, M / 64), 256, 0, stream>>>(
        xe_hi, xe_lo, Wih_e, nullptr, nullptr, bih_e, xg, E_, G4H);
  }
  for (int t = 0; t < TSEQ; ++t) {
    const __bf16* hph = t ? hsh_e + (size_t)(t - 1) * B_ * H_ : h0hi;
    const __bf16* hpl = t ? hsl_e + (size_t)(t - 1) * B_ * H_ : h0lo;
    const float*  cp  = t ? cs_e + (size_t)(t - 1) * B_ * H_ : c0;
    if (big)
      lstm_step_bf16x3<true><<<H_ / 16, 256, 0, stream>>>(hph, hpl, H_, cp,
          nullptr, whhh, whhl, bhh_e, xg + (size_t)t * B_ * G4H,
          hs_e + (size_t)t * B_ * H_,
          hsh_e + (size_t)t * B_ * H_, hsl_e + (size_t)t * B_ * H_, H_,
          cs_e + (size_t)t * B_ * H_);
    else
      lstm_step_bf16x3<false><<<H_ / 16, 256, 0, stream>>>(hph, hpl, H_, cp,
          Whh_e, nullptr, nullptr, bhh_e, xg + (size_t)t * B_ * G4H,
          hs_e + (size_t)t * B_ * H_,
          hsh_e + (size_t)t * B_ * H_, hsl_e + (size_t)t * B_ * H_, H_,
          cs_e + (size_t)t * B_ * H_);
  }
  gather_last<<<(B_ * H_) / 256, 256, 0, stream>>>(x_mask, hs_e, cs_e, hinhi, hinlo, cinit);

  // ---- decoder ----
  embed_rows_hl<<<M, 256, 0, stream>>>(emb_cn, y, xe_hi, xe_lo);
  if (big) {
    split_w<<<NIH / 256, 256, 0, stream>>>(Wih_d, wihh, wihl, NIH);   // reuse buffers
    split_w<<<NHH / 256, 256, 0, stream>>>(Whh_d, whhh, whhl, NHH);
    gemm_bf16x3<true><<<dim3(G4H / 256, M / 64), 256, 0, stream>>>(
        xe_hi, xe_lo, nullptr, wihh, wihl, bih_d, xg, E_, G4H);
  } else {
    gemm_bf16x3<false><<<dim3(G4H / 256, M / 64), 256, 0, stream>>>(
        xe_hi, xe_lo, Wih_d, nullptr, nullptr, bih_d, xg, E_, G4H);
  }
  for (int t = 0; t < TSEQ; ++t) {
    const __bf16* hph; const __bf16* hpl; int hps;
    if (t == 0) { hph = hinhi; hpl = hinlo; hps = H_; }
    else        { hph = hdh + (size_t)(t - 1) * H_; hpl = hdl + (size_t)(t - 1) * H_; hps = TSEQ * H_; }
    const float* cp = t ? (((t - 1) & 1) ? cb1 : cb0) : cinit;
    float* co = (t & 1) ? cb1 : cb0;
    if (big)
      lstm_step_bf16x3<true><<<H_ / 16, 256, 0, stream>>>(hph, hpl, hps, cp,
          nullptr, whhh, whhl, bhh_d, xg + (size_t)t * B_ * G4H,
          hidd + (size_t)t * H_,
          hdh + (size_t)t * H_, hdl + (size_t)t * H_, TSEQ * H_, co);
    else
      lstm_step_bf16x3<false><<<H_ / 16, 256, 0, stream>>>(hph, hpl, hps, cp,
          Whh_d, nullptr, nullptr, bhh_d, xg + (size_t)t * B_ * G4H,
          hidd + (size_t)t * H_,
          hdh + (size_t)t * H_, hdl + (size_t)t * H_, TSEQ * H_, co);
  }

  // ---- output projection + log_softmax ----
  if (big) {
    split_w<<<NOU / 256, 256, 0, stream>>>(Wout, wouth, woutl, NOU);
    gemm_bf16x3<true><<<dim3(VOC / 256, M / 64), 256, 0, stream>>>(
        hdh, hdl, nullptr, wouth, woutl, bout, decoded, H_, VOC);
  } else {
    gemm_bf16x3<false><<<dim3(VOC / 256, M / 64), 256, 0, stream>>>(
        hdh, hdl, Wout, nullptr, nullptr, bout, decoded, H_, VOC);
  }
  log_softmax_rows<<<M, 256, 0, stream>>>(decoded);
}
